// SplitHeadSAMDecoderAttention_7387343749257
// MI455X (gfx1250) — matrix operand order repaired and checked by reading, not compiled, not run on hardware
//
#include <hip/hip_runtime.h>
#include <hip/hip_bf16.h>

// SplitHead SAM decoder attention for gfx1250 (MI455X).
// Pipeline: [proj q/k/v -> bf16 (V transposed, log2e/sqrt(dh) folded into Q)]
// -> [flash attention, bf16 WMMA, fp32 online softmax in exp2 domain]
// -> [out-proj bf16 WMMA -> fp32].  Needs 64 MiB of d_ws scratch.

#define BB   8
#define NQ   4096
#define NK   4096
#define CDIM 256
#define NH   8
#define DH   32

typedef __bf16 v16bf __attribute__((ext_vector_type(16)));
typedef __bf16 v8bf  __attribute__((ext_vector_type(8)));
typedef float  v8f   __attribute__((ext_vector_type(8)));

__device__ __forceinline__ __bf16 f2bf(float x) { return (__bf16)x; }

__device__ __forceinline__ v16bf cat8(v8bf lo, v8bf hi) {
  return __builtin_shufflevector(lo, hi, 0,1,2,3,4,5,6,7,8,9,10,11,12,13,14,15);
}

__device__ __forceinline__ v8f wmma_bf16(v16bf a, v16bf b, v8f c) {
  // (neg_a, A, neg_b, B, c_mod, C, reuse_a, reuse_b)
  return __builtin_amdgcn_wmma_f32_16x16x32_bf16(false, a, false, b, (short)0, c,
                                                 false, false);
}

// ---------------------------------------------------------------------------
// Kernel 1: per-head projections as one GEMM each: [32768,256] x [256,256]^T.
// blockIdx.z selects q/k/v. Output bf16; V written transposed [b,h][d][n];
// log2(e)/sqrt(dh) folded into Q so the flash kernel can use raw v_exp_f32.
// ---------------------------------------------------------------------------
__global__ __launch_bounds__(128) void proj_kernel(
    const float* __restrict__ q, const float* __restrict__ k,
    const float* __restrict__ v,
    const float* __restrict__ Wq, const float* __restrict__ bq,
    const float* __restrict__ Wk, const float* __restrict__ bk,
    const float* __restrict__ Wv, const float* __restrict__ bv,
    __bf16* __restrict__ Qh, __bf16* __restrict__ Kh, __bf16* __restrict__ Vt)
{
  const int kind = blockIdx.z;
  const float* X    = (kind == 0) ? q  : (kind == 1) ? k  : v;
  const float* W    = (kind == 0) ? Wq : (kind == 1) ? Wk : Wv;
  const float* bias = (kind == 0) ? bq : (kind == 1) ? bk : bv;

  const int lane = threadIdx.x & 31;
  const int wave = threadIdx.x >> 5;
  const int sub  = lane >> 4;   // K-half select: contiguous k = 16*sub + s in both A and B
  const int l16  = lane & 15;   // row (A) / column (B,C/D)

  const int mBase = blockIdx.x * 64 + wave * 16;
  const int nBase = blockIdx.y * 16;
  const int arow  = mBase + l16;
  const int brow  = nBase + l16;   // output channel = h*32+d

  v8f acc = {};
  #pragma unroll
  for (int kb = 0; kb < CDIM; kb += 32) {
    const float* ap = X + (size_t)arow * CDIM + kb + sub * 16;
    const float* bp = W + (size_t)brow * CDIM + kb + sub * 16;
    v16bf afrag, bfrag;
    #pragma unroll
    for (int i = 0; i < 8; ++i) { afrag[i] = f2bf(ap[i]); afrag[8 + i] = f2bf(ap[8 + i]); }
    #pragma unroll
    for (int i = 0; i < 16; ++i) bfrag[i] = f2bf(bp[i]);
    acc = wmma_bf16(afrag, bfrag, acc);
  }

  const float bval = bias[brow];
  // log2(e) / sqrt(32): flash softmax runs in the exp2 domain.
  const float scl = (kind == 0) ? 0.25503526f : 1.0f;
  const int h = brow >> 5;
  const int d = brow & 31;
  #pragma unroll
  for (int r = 0; r < 8; ++r) {
    const int m   = mBase + r + 8 * sub;   // C/D layout: M = r + 8*(lane/16)
    const int b   = m >> 12;
    const int tok = m & 4095;
    const float val = (acc[r] + bval) * scl;
    if (kind == 2)
      Vt[(((size_t)b * NH + h) * DH + d) * NK + tok] = f2bf(val);
    else if (kind == 0)
      Qh[(((size_t)b * NH + h) * NQ + tok) * DH + d] = f2bf(val);
    else
      Kh[(((size_t)b * NH + h) * NK + tok) * DH + d] = f2bf(val);
  }
}

// ---------------------------------------------------------------------------
// Kernel 2: flash attention. 8 waves/block, 16 query rows per wave (one
// 16x32 Q A-fragment covers the whole head dim). Key loop in tiles of 64:
// 4 S WMMAs, one online-softmax update (shfl_xor butterflies amortized over
// 64 keys, exp2 via v_exp_f32), P transposed C-layout -> A-layout through a
// per-wave padded LDS tile, then 4 accumulating PV WMMAs. K/V fragments are
// identical across the 8 waves -> deduped by WGP$, sourced from L2
// (48 MB of bf16 tensors << 192 MB L2).
// ---------------------------------------------------------------------------
__global__ __launch_bounds__(256) void flash_kernel(
    const __bf16* __restrict__ Qh, const __bf16* __restrict__ Kh,
    const __bf16* __restrict__ Vt, __bf16* __restrict__ Aout)
{
  __shared__ __bf16 plds[8][16][72];   // per-wave 16x64 P tile, padded stride

  const int lane = threadIdx.x & 31;
  const int wave = threadIdx.x >> 5;
  const int sub  = lane >> 4;
  const int l16  = lane & 15;

  const int bh    = blockIdx.y;                 // b*NH + h
  const int qBase = blockIdx.x * 128 + wave * 16;

  const __bf16* Q = Qh + (size_t)bh * NQ * DH;
  const __bf16* K = Kh + (size_t)bh * NK * DH;
  const __bf16* V = Vt + (size_t)bh * DH * NK;

  v16bf qf;
  {
    const __bf16* qp = Q + (size_t)(qBase + l16) * DH + sub * 16;
    qf = cat8(*(const v8bf*)qp, *(const v8bf*)(qp + 8));
  }

  v8f acc0 = {}, acc1 = {};
  float rmax[8], rsum[8];
  #pragma unroll
  for (int r = 0; r < 8; ++r) { rmax[r] = -1e30f; rsum[r] = 0.f; }

  for (int kb = 0; kb < NK; kb += 64) {
    // S = Q K^T for 64 keys: four 16x16 tiles.
    v8f s[4];
    #pragma unroll
    for (int t = 0; t < 4; ++t) {
      v16bf kf = *(const v16bf*)(K + (size_t)(kb + 16 * t + l16) * DH + sub * 16);
      v8f z = {};
      s[t] = wmma_bf16(qf, kf, z);
    }

    // Online softmax update (exp2 domain), one pass per 64 keys.
    #pragma unroll
    for (int r = 0; r < 8; ++r) {
      float tm = fmaxf(fmaxf(s[0][r], s[1][r]), fmaxf(s[2][r], s[3][r]));
      #pragma unroll
      for (int off = 1; off < 16; off <<= 1) tm = fmaxf(tm, __shfl_xor(tm, off, 32));
      const float mnew = fmaxf(rmax[r], tm);
      const float cf   = __builtin_amdgcn_exp2f(rmax[r] - mnew);
      float ps = 0.f;
      #pragma unroll
      for (int t = 0; t < 4; ++t) {
        const float p = __builtin_amdgcn_exp2f(s[t][r] - mnew);
        ps += p;
        // C-layout -> LDS: row = r + 8*(lane/16), col = key within 64-tile
        plds[wave][r + 8 * sub][16 * t + l16] = f2bf(p);
      }
      #pragma unroll
      for (int off = 1; off < 16; off <<= 1) ps += __shfl_xor(ps, off, 32);
      rsum[r] = rsum[r] * cf + ps;
      rmax[r] = mnew;
      acc0[r] *= cf;
      acc1[r] *= cf;
    }

    asm volatile("s_wait_dscnt 0" ::: "memory");

    // PV: two P A-fragments (keys 0..31 / 32..63 of the tile) x two dh halves.
    #pragma unroll
    for (int kh = 0; kh < 2; ++kh) {
      const __bf16* pp = &plds[wave][l16][kh * 32 + sub * 16];
      v16bf pf = cat8(*(const v8bf*)pp, *(const v8bf*)(pp + 8));
      v16bf vf0 = *(const v16bf*)(V + (size_t)l16 * NK        + kb + kh * 32 + sub * 16);
      v16bf vf1 = *(const v16bf*)(V + (size_t)(16 + l16) * NK + kb + kh * 32 + sub * 16);
      acc0 = wmma_bf16(pf, vf0, acc0);
      acc1 = wmma_bf16(pf, vf1, acc1);
    }
  }

  const int b = bh >> 3;
  const int h = bh & 7;
  #pragma unroll
  for (int r = 0; r < 8; ++r) {
    const int qrow  = qBase + r + 8 * sub;
    const float inv = 1.0f / rsum[r];
    __bf16* dst = Aout + ((size_t)b * NQ + qrow) * (NH * DH) + h * DH;
    dst[l16]      = f2bf(acc0[r] * inv);
    dst[16 + l16] = f2bf(acc1[r] * inv);
  }
}

// ---------------------------------------------------------------------------
// Kernel 3: out-projection GEMM [32768,256] x Wo^T + bo, fp32 output.
// ---------------------------------------------------------------------------
__global__ __launch_bounds__(128) void outproj_kernel(
    const __bf16* __restrict__ Aout, const float* __restrict__ Wo,
    const float* __restrict__ bo, float* __restrict__ Y)
{
  const int lane = threadIdx.x & 31;
  const int wave = threadIdx.x >> 5;
  const int sub  = lane >> 4;
  const int l16  = lane & 15;

  const int mBase = blockIdx.x * 64 + wave * 16;
  const int nBase = blockIdx.y * 16;

  v8f acc = {};
  #pragma unroll
  for (int kb = 0; kb < CDIM; kb += 32) {
    const __bf16* ap = Aout + (size_t)(mBase + l16) * CDIM + kb + sub * 16;
    v16bf afrag = cat8(*(const v8bf*)ap, *(const v8bf*)(ap + 8));
    const float* bp = Wo + (size_t)(nBase + l16) * CDIM + kb + sub * 16;
    v16bf bfrag;
    #pragma unroll
    for (int i = 0; i < 16; ++i) bfrag[i] = f2bf(bp[i]);
    acc = wmma_bf16(afrag, bfrag, acc);
  }

  const float bval = bo[nBase + l16];
  #pragma unroll
  for (int r = 0; r < 8; ++r) {
    const int m = mBase + r + 8 * sub;
    Y[(size_t)m * CDIM + nBase + l16] = acc[r] + bval;
  }
}

// ---------------------------------------------------------------------------
extern "C" void kernel_launch(void* const* d_in, const int* in_sizes, int n_in,
                              void* d_out, int out_size, void* d_ws, size_t ws_size,
                              hipStream_t stream) {
  (void)in_sizes; (void)n_in; (void)out_size;
  const float* q  = (const float*)d_in[0];
  const float* k  = (const float*)d_in[1];
  const float* v  = (const float*)d_in[2];
  const float* Wq = (const float*)d_in[3];
  const float* bq = (const float*)d_in[4];
  const float* Wk = (const float*)d_in[5];
  const float* bk = (const float*)d_in[6];
  const float* Wv = (const float*)d_in[7];
  const float* bv = (const float*)d_in[8];
  const float* Wo = (const float*)d_in[9];
  const float* bo = (const float*)d_in[10];

  const size_t nTok   = (size_t)BB * NQ;        // 32768 rows
  const size_t perTen = nTok * NH * DH;         // 8.39M bf16 elems = 16 MiB
  if (ws_size < 4 * perTen * sizeof(__bf16)) return;  // need 64 MiB

  __bf16* Qh = (__bf16*)d_ws;
  __bf16* Kh = Qh + perTen;
  __bf16* Vt = Kh + perTen;
  __bf16* Ao = Vt + perTen;

  proj_kernel<<<dim3((unsigned)(nTok / 64), CDIM / 16, 3), 128, 0, stream>>>(
      q, k, v, Wq, bq, Wk, bk, Wv, bv, Qh, Kh, Vt);
  flash_kernel<<<dim3(NQ / 128, BB * NH), 256, 0, stream>>>(Qh, Kh, Vt, Ao);
  outproj_kernel<<<dim3((unsigned)(nTok / 64), CDIM / 16), 128, 0, stream>>>(
      Ao, Wo, bo, (float*)d_out);
}
